// GNNLayer_80092550136106
// MI455X (gfx1250) — matrix operand order repaired and checked by reading, not compiled, not run on hardware
//
#include <hip/hip_runtime.h>
#include <hip/hip_bf16.h>

typedef __attribute__((ext_vector_type(16))) __bf16 v16bf;
typedef __attribute__((ext_vector_type(8)))  __bf16 v8bf;
typedef __attribute__((ext_vector_type(4)))  __bf16 v4bf;
typedef __attribute__((ext_vector_type(8)))  float  v8f;
typedef __attribute__((ext_vector_type(2)))  float  v2f;

#define XST 264   // X row stride in bf16 (256 + 8 pad -> 4-bank rotate per row)
#define WST 264   // Wcat row stride in bf16
#define HST 68    // hs*hr row stride in bf16 (64 + 4 pad, keeps 8B alignment)
#define TILES_PER_BLOCK 16

static __device__ inline v8f vzero8() {
    v8f z;
#pragma unroll
    for (int i = 0; i < 8; ++i) z[i] = 0.0f;
    return z;
}

static __device__ inline v4bf cvt4(float x, float y, float z, float w) {
    v4bf r;
    r[0] = (__bf16)x; r[1] = (__bf16)y; r[2] = (__bf16)z; r[3] = (__bf16)w;
    return r;
}

// A fragment: 16x32 bf16 tile. lanes 0-15: row M=lr, K = kc*32 + {0..7, 16..23}
//             lanes 16-31: row M=lr, K = kc*32 + {8..15, 24..31}
static __device__ inline v16bf loadA(const __bf16* X, int mt, int kc, int lr, int hf) {
    const __bf16* row = X + (mt * 16 + lr) * XST + kc * 32 + hf * 8;
    v8bf lo = *(const v8bf*)(row);
    v8bf hi = *(const v8bf*)(row + 16);
    return __builtin_shufflevector(lo, hi, 0,1,2,3,4,5,6,7,8,9,10,11,12,13,14,15);
}

// B fragment: 32x16 bf16 tile. lane n holds column N=lr; lanes 0-15: K=kc*32+{0..7, 16..23},
// lanes 16-31: K=kc*32+{8..15, 24..31}. B[k][n] = Wcat[n][k] -> two 8-bf16 runs of row n.
static __device__ inline v16bf loadB(const __bf16* W, int nt, int kc, int lr, int hf) {
    const __bf16* row = W + (nt * 16 + lr) * WST + kc * 32 + hf * 8;
    v8bf lo = *(const v8bf*)(row);
    v8bf hi = *(const v8bf*)(row + 16);
    return __builtin_shufflevector(lo, hi, 0,1,2,3,4,5,6,7,8,9,10,11,12,13,14,15);
}

__global__ void zero_ws_kernel(float* __restrict__ p, long n) {
    long i = (long)blockIdx.x * blockDim.x + threadIdx.x;
    long stride = (long)gridDim.x * blockDim.x;
    for (; i < n; i += stride) p[i] = 0.0f;
}

__global__ __launch_bounds__(32)
void edge_kernel(const int*   __restrict__ edges,
                 const float* __restrict__ hidden,
                 const float* __restrict__ rela,
                 const float* __restrict__ qemb,
                 const float* __restrict__ WsW, const float* __restrict__ Wsb,
                 const float* __restrict__ WrW, const float* __restrict__ WqW,
                 const float* __restrict__ WqrW,
                 const float* __restrict__ waW, const float* __restrict__ wab,
                 float* __restrict__ acc_ws,     // N*64 f32 accumulator
                 float* __restrict__ alpha_out,  // E
                 int E)
{
    __shared__ __bf16 Wlds[64 * WST];   // 33792 B: Wcat bf16, rows = attn dim a
    __shared__ __bf16 Xlds[32 * XST];   // 16896 B: [hs|hr|hqr|hr*hqr] per edge, bf16
    __shared__ __bf16 HSHR[32 * HST];   //  4352 B: hs*hr per edge, bf16
    __shared__ float  alpha_s[32];
    __shared__ int    obj_s[32];

    const int lane = threadIdx.x;       // one wave32 per block
    const int lr = lane & 15;
    const int hf = lane >> 4;

    // ---- stage concatenated weights [Ws|Wr|Wq|Wqr] (rows a=0..63, K=0..255) as bf16 ----
#pragma unroll
    for (int seg = 0; seg < 4; ++seg) {
        const float* W = (seg == 0) ? WsW : (seg == 1) ? WrW : (seg == 2) ? WqW : WqrW;
        for (int base = 0; base < 64 * 64; base += 128) {
            int idx = base + lane * 4;
            float4 w = *(const float4*)(W + idx);
            int a = idx >> 6, c = idx & 63;
            *(v4bf*)&Wlds[a * WST + seg * 64 + c] = cvt4(w.x, w.y, w.z, w.w);
        }
    }
    __syncthreads();

    const long tile0 = (long)blockIdx.x * TILES_PER_BLOCK;
    for (int t = 0; t < TILES_PER_BLOCK; ++t) {
        long tile  = tile0 + t;
        long ebase = tile * 32;
        if (ebase >= E) break;
        int count = E - (int)ebase;
        if (count > 32) count = 32;

        // ---- gather phase: lane i owns edge ebase+i ----
        if (lane < count) {
            long e = ebase + lane;
            int q = edges[e * 6 + 0];
            int r = edges[e * 6 + 2];
            int s = edges[e * 6 + 4];
            int o = edges[e * 6 + 5];
            obj_s[lane] = o;
            const float4* hsp = (const float4*)(hidden + (long)s * 64);
            const float4* hrp = (const float4*)(rela   + (long)r * 64);
            const float4* hqp = (const float4*)(qemb   + (long)q * 64);
#pragma unroll
            for (int j = 0; j < 16; ++j) {
                float4 hs = hsp[j], hr = hrp[j], hq = hqp[j];
                __bf16* xr = &Xlds[lane * XST];
                *(v4bf*)(xr +       4 * j) = cvt4(hs.x, hs.y, hs.z, hs.w);
                *(v4bf*)(xr +  64 + 4 * j) = cvt4(hr.x, hr.y, hr.z, hr.w);
                *(v4bf*)(xr + 128 + 4 * j) = cvt4(hq.x, hq.y, hq.z, hq.w);
                *(v4bf*)(xr + 192 + 4 * j) = cvt4(hr.x * hq.x, hr.y * hq.y,
                                                  hr.z * hq.z, hr.w * hq.w);
                *(v4bf*)&HSHR[lane * HST + 4 * j] = cvt4(hs.x * hr.x, hs.y * hr.y,
                                                         hs.z * hr.z, hs.w * hr.w);
            }
        } else {
            obj_s[lane] = 0;
            v4bf z = cvt4(0.f, 0.f, 0.f, 0.f);
#pragma unroll
            for (int j = 0; j < 256; j += 4) *(v4bf*)&Xlds[lane * XST + j] = z;
#pragma unroll
            for (int j = 0; j < 64; j += 4) *(v4bf*)&HSHR[lane * HST + j] = z;
        }
        __syncthreads();

        // ---- pre = X @ Wcat^T via bf16 WMMA, fp32 accumulate ----
        v8f acc[2][4];
#pragma unroll
        for (int mt = 0; mt < 2; ++mt)
#pragma unroll
            for (int nt = 0; nt < 4; ++nt) acc[mt][nt] = vzero8();

#pragma unroll
        for (int kc = 0; kc < 8; ++kc) {
            v16bf A0 = loadA(Xlds, 0, kc, lr, hf);
            v16bf A1 = loadA(Xlds, 1, kc, lr, hf);
#pragma unroll
            for (int nt = 0; nt < 4; ++nt) {
                v16bf B = loadB(Wlds, nt, kc, lr, hf);
                acc[0][nt] = __builtin_amdgcn_wmma_f32_16x16x32_bf16(
                    false, A0, false, B, (short)0, acc[0][nt], false, false);
                acc[1][nt] = __builtin_amdgcn_wmma_f32_16x16x32_bf16(
                    false, A1, false, B, (short)0, acc[1][nt], false, false);
            }
        }

        // ---- bias + relu + wa-dot (half-wave shfl reduce) + sigmoid -> alpha ----
        float bias[4], wav[4];
#pragma unroll
        for (int nt = 0; nt < 4; ++nt) {
            bias[nt] = Wsb[nt * 16 + lr];
            wav[nt]  = waW[nt * 16 + lr];
        }
        float wb = wab[0];

#pragma unroll
        for (int mt = 0; mt < 2; ++mt) {
            float p[8];
#pragma unroll
            for (int v = 0; v < 8; ++v) p[v] = 0.0f;
#pragma unroll
            for (int nt = 0; nt < 4; ++nt)
#pragma unroll
                for (int v = 0; v < 8; ++v)
                    p[v] += fmaxf(acc[mt][nt][v] + bias[nt], 0.0f) * wav[nt];
            // reduce across the 16 lanes of each half (each lane holds one N)
#pragma unroll
            for (int mask = 1; mask < 16; mask <<= 1)
#pragma unroll
                for (int v = 0; v < 8; ++v) p[v] += __shfl_xor(p[v], mask, 32);
            if (lr == 0) {
#pragma unroll
                for (int v = 0; v < 8; ++v) {
                    int m = mt * 16 + hf * 8 + v;  // edge within tile
                    float al = 1.0f / (1.0f + __expf(-(p[v] + wb)));
                    alpha_s[m] = al;
                    if (m < count) alpha_out[ebase + m] = al;
                }
            }
        }
        __syncthreads();

        // ---- message scatter: alpha * (hs*hr) -> atomic segment sum ----
        for (int m = 0; m < count; ++m) {
            float al = alpha_s[m];
            long o = obj_s[m];
            float v0 = al * (float)HSHR[m * HST + 2 * lane];
            float v1 = al * (float)HSHR[m * HST + 2 * lane + 1];
            atomicAdd(&acc_ws[o * 64 + 2 * lane], v0);
            atomicAdd(&acc_ws[o * 64 + 2 * lane + 1], v1);
        }
        __syncthreads();
    }
}

// hidden_new = message_agg (N x 64, f32) @ Wh^T via V_WMMA_F32_16X16X4_F32
__global__ __launch_bounds__(256)
void out_gemm_kernel(const float* __restrict__ acc_ws,
                     const float* __restrict__ WhW,
                     float* __restrict__ out, int N)
{
    const int wave = threadIdx.x >> 5;
    const int lane = threadIdx.x & 31;
    const int lr = lane & 15;
    const int hf = lane >> 4;
    const long rbase = ((long)blockIdx.x * 8 + wave) * 16;
    if (rbase >= N) return;

    v8f acc[4];
#pragma unroll
    for (int nt = 0; nt < 4; ++nt) acc[nt] = vzero8();

    const long arow = rbase + lr;
    const bool rowok = arow < N;
    const float* ap = acc_ws + arow * 64;

#pragma unroll
    for (int k = 0; k < 64; k += 4) {
        v2f A;
        if (rowok) {
            float2 a2 = *(const float2*)(ap + k + hf * 2);
            A[0] = a2.x; A[1] = a2.y;
        } else {
            A[0] = 0.0f; A[1] = 0.0f;
        }
#pragma unroll
        for (int nt = 0; nt < 4; ++nt) {
            float2 b2 = *(const float2*)(WhW + (nt * 16 + lr) * 64 + k + hf * 2);
            v2f B; B[0] = b2.x; B[1] = b2.y;
            acc[nt] = __builtin_amdgcn_wmma_f32_16x16x4_f32(
                false, A, false, B, (short)0, acc[nt], false, false);
        }
    }

#pragma unroll
    for (int nt = 0; nt < 4; ++nt)
#pragma unroll
        for (int v = 0; v < 8; ++v) {
            long row = rbase + hf * 8 + v;
            if (row < N) out[row * 64 + nt * 16 + lr] = acc[nt][v];
        }
}

extern "C" void kernel_launch(void* const* d_in, const int* in_sizes, int n_in,
                              void* d_out, int out_size, void* d_ws, size_t ws_size,
                              hipStream_t stream) {
    const float* q_emb  = (const float*)d_in[1];
    const float* rela   = (const float*)d_in[2];
    const float* hidden = (const float*)d_in[3];
    const int*   edges  = (const int*)d_in[4];
    const float* WsW  = (const float*)d_in[6];
    const float* Wsb  = (const float*)d_in[7];
    const float* WrW  = (const float*)d_in[8];
    const float* WqW  = (const float*)d_in[9];
    const float* WqrW = (const float*)d_in[10];
    const float* waW  = (const float*)d_in[11];
    const float* wab  = (const float*)d_in[12];
    const float* WhW  = (const float*)d_in[13];

    const int E = in_sizes[4] / 6;
    const int N = in_sizes[3] / 64;

    float* ws         = (float*)d_ws;
    float* out_hidden = (float*)d_out;
    float* out_alpha  = out_hidden + (long)N * 64;

    zero_ws_kernel<<<1024, 256, 0, stream>>>(ws, (long)N * 64);

    const int ntiles  = (E + 31) / 32;
    const int nblocks = (ntiles + TILES_PER_BLOCK - 1) / TILES_PER_BLOCK;
    edge_kernel<<<nblocks, 32, 0, stream>>>(edges, hidden, rela, q_emb,
                                            WsW, Wsb, WrW, WqW, WqrW, waW, wab,
                                            ws, out_alpha, E);

    const int rtiles = (N + 15) / 16;
    out_gemm_kernel<<<(rtiles + 7) / 8, 256, 0, stream>>>(ws, WhW, out_hidden, N);
}
